// BahdanauAttention_33243046871455
// MI455X (gfx1250) — matrix operand order repaired and checked by reading, not compiled, not run on hardware
//
#include <hip/hip_runtime.h>

// ---------------------------------------------------------------------------
// Bahdanau attention, MI455X (gfx1250, wave32, WMMA).
// B=256, P=64, F=2048, U=1024.
//   proj_f = features @ W1 + b1            (16384 x 2048 x 1024 GEMM, bf16 WMMA)
//   score  = tanh(proj_f + proj_h)         (fused epilogue, never materialized)
//   logits = score @ V_w                   (fused cross-lane reduce)
//   attn   = softmax_P(logits); context = sum_p attn * features
// Workspace layout (needs 7.5 MB):
//   [0,4MB)        W1T  bf16 [1024][2048]
//   [4MB,6MB)      W2T  bf16 [1024][1024]
//   [6MB,6.5MB)    hidden bf16 [256][1024]
//   [6.5MB,7.5MB)  proj_h f32 [256][1024]
// ---------------------------------------------------------------------------

typedef __attribute__((ext_vector_type(16))) __bf16 v16bf;
typedef __attribute__((ext_vector_type(8)))  float  v8f;

union ABFrag { uint4 q[2]; v16bf v; };   // 32 bytes = 16 bf16 per lane

__device__ __forceinline__ unsigned short f2bf(float f) {
  unsigned u = __builtin_bit_cast(unsigned, f);
  u += 0x7FFFu + ((u >> 16) & 1u);       // round-to-nearest-even
  return (unsigned short)(u >> 16);
}

// in: f32 row-major [K][N]  ->  out: bf16 row-major [N][K]
__global__ __launch_bounds__(256)
void k_transpose_bf16(const float* __restrict__ in, unsigned short* __restrict__ out,
                      int K, int N) {
  __shared__ unsigned short t[32][33];
  const int k0 = blockIdx.x * 32, n0 = blockIdx.y * 32;
  for (int i = threadIdx.y; i < 32; i += 8)
    t[i][threadIdx.x] = f2bf(in[(size_t)(k0 + i) * N + n0 + threadIdx.x]);
  __syncthreads();
  for (int i = threadIdx.y; i < 32; i += 8)
    out[(size_t)(n0 + i) * K + k0 + threadIdx.x] = t[threadIdx.x][i];
}

__global__ __launch_bounds__(256)
void k_f32_to_bf16(const float* __restrict__ in, unsigned short* __restrict__ out, int n) {
  int i = blockIdx.x * blockDim.x + threadIdx.x;
  if (i < n) out[i] = f2bf(in[i]);
}

// proj_h = hidden @ W2 + b2 : one wave per 16x16 tile, K=1024.
__global__ __launch_bounds__(128)
void k_proj_h(const unsigned short* __restrict__ hb,   // bf16 [256][1024]
              const unsigned short* __restrict__ w2t,  // bf16 [1024][1024] (n-major)
              const float* __restrict__ w2b,
              float* __restrict__ ph) {                // f32 [256][1024]
  const int tid = threadIdx.x, w = tid >> 5, lane = tid & 31;
  const int l15 = lane & 15, lh = lane >> 4;
  const int m_tile = blockIdx.x, n_tile = blockIdx.y * 4 + w;
  const unsigned short* arow = hb  + (size_t)(m_tile * 16 + l15) * 1024;
  const unsigned short* bcol = w2t + (size_t)(n_tile * 16 + l15) * 1024;
  v8f acc = {};
  for (int k = 0; k < 1024; k += 32) {
    ABFrag a, bf_;
    const int kA = k + (lh << 3);        // A: K 0..7 / 16..23 (+8 for hi lanes)
    a.q[0]   = *(const uint4*)(arow + kA);
    a.q[1]   = *(const uint4*)(arow + kA + 16);
    const int kB = k + (lh << 3);        // B: K 0..7 / 16..23 (+8 for hi lanes)
    bf_.q[0] = *(const uint4*)(bcol + kB);
    bf_.q[1] = *(const uint4*)(bcol + kB + 16);
    acc = __builtin_amdgcn_wmma_f32_16x16x32_bf16(false, a.v, false, bf_.v,
                                                  (short)0, acc, false, false);
  }
  const int col = n_tile * 16 + l15;
  const float bias = w2b[col];
#pragma unroll
  for (int r = 0; r < 8; ++r) {
    const int row = m_tile * 16 + r + (lh << 3);   // C: VGPR r -> M=r (+8 hi lanes)
    ph[(size_t)row * 1024 + col] = acc[r] + bias;
  }
}

// Fused: features[b] -> LDS bf16 -> WMMA GEMM -> tanh/V reduce -> softmax -> context
__global__ __launch_bounds__(512)
void k_attention(const float* __restrict__ features,       // f32 [256][64][2048]
                 const unsigned short* __restrict__ w1t,   // bf16 [1024][2048]
                 const float* __restrict__ w1b,            // [1024]
                 const float* __restrict__ vw,             // [1024]
                 const float* __restrict__ ph,             // f32 [256][1024]
                 float* __restrict__ out_ctx,              // f32 [256][2048]
                 float* __restrict__ out_attn) {           // f32 [256][64]
  extern __shared__ unsigned char smem_raw[];
  unsigned short* sA = (unsigned short*)smem_raw;            // 64*2048 bf16 = 256 KB
  float* sLogits = (float*)(smem_raw + 64 * 2048 * 2);       // 64
  float* sAttn   = sLogits + 64;                             // 64

  const int b = blockIdx.x;
  const int tid = threadIdx.x;
  const int w = tid >> 5, lane = tid & 31;
  const int l15 = lane & 15, lh = lane >> 4;

  if (tid < 64) sLogits[tid] = 0.0f;

  // ---- Phase 1: stage features[b] as bf16 in LDS (only GEMM-pass HBM read) ----
  const float4* f4 = (const float4*)(features + (size_t)b * 64 * 2048);
  uint2* sA2 = (uint2*)sA;
  for (int i = tid; i < 64 * 2048 / 4; i += 512) {
    float4 v = f4[i];
    uint2 pkd;
    pkd.x = (unsigned)f2bf(v.x) | ((unsigned)f2bf(v.y) << 16);
    pkd.y = (unsigned)f2bf(v.z) | ((unsigned)f2bf(v.w) << 16);
    sA2[i] = pkd;
  }
  __syncthreads();

  // ---- Phase 2: 64x1024 GEMM over K=2048 + fused tanh/V_w logit reduction ----
  // 16 waves x 4 n_tiles; each wave keeps all 4 m_tiles live so one B fragment
  // feeds 4 WMMAs (B stays in WGP$/L2, A streams from LDS).
  for (int j = 0; j < 4; ++j) {
    const int n_tile = w + 16 * j;
    const int ncol = n_tile * 16 + l15;
    const unsigned short* bcol = w1t + (size_t)ncol * 2048;
    v8f acc[4] = {{}, {}, {}, {}};
    for (int k = 0; k < 2048; k += 32) {
      ABFrag bf_;
      const int kB = k + (lh << 3);
      bf_.q[0] = *(const uint4*)(bcol + kB);
      bf_.q[1] = *(const uint4*)(bcol + kB + 16);
      const int kA = k + (lh << 3);
#pragma unroll
      for (int m = 0; m < 4; ++m) {
        ABFrag a;
        const unsigned short* arow = sA + (size_t)(m * 16 + l15) * 2048 + kA;
        a.q[0] = *(const uint4*)(arow);
        a.q[1] = *(const uint4*)(arow + 16);
        acc[m] = __builtin_amdgcn_wmma_f32_16x16x32_bf16(false, a.v, false, bf_.v,
                                                         (short)0, acc[m], false, false);
      }
    }
    // epilogue: logits[p] += sum_n tanh(proj_f + proj_h + b1) * V_w[n]
    const float sadd = ph[(size_t)b * 1024 + ncol] + w1b[ncol];
    const float vwn = vw[ncol];
#pragma unroll
    for (int m = 0; m < 4; ++m) {
#pragma unroll
      for (int r = 0; r < 8; ++r) {
        float t = tanhf(acc[m][r] + sadd) * vwn;
        t += __shfl_xor(t, 1);  t += __shfl_xor(t, 2);
        t += __shfl_xor(t, 4);  t += __shfl_xor(t, 8);   // sum over 16 N-lanes
        if (l15 == 0)
          atomicAdd(&sLogits[m * 16 + r + (lh << 3)], t); // ds_add_f32
      }
    }
  }
  __syncthreads();

  // ---- Phase 3: softmax over P=64 (V_b is uniform -> softmax-invariant) ----
  if (w == 0) {
    float a = sLogits[lane], c = sLogits[lane + 32];
    float mx = fmaxf(a, c);
#pragma unroll
    for (int off = 16; off; off >>= 1) mx = fmaxf(mx, __shfl_xor(mx, off));
    float ea = __expf(a - mx), ec = __expf(c - mx);
    float s = ea + ec;
#pragma unroll
    for (int off = 16; off; off >>= 1) s += __shfl_xor(s, off);
    const float inv = 1.0f / s;
    sAttn[lane] = ea * inv;      sAttn[lane + 32] = ec * inv;
    out_attn[b * 64 + lane] = ea * inv;
    out_attn[b * 64 + lane + 32] = ec * inv;
  }
  __syncthreads();

  // ---- Phase 4: context = sum_p attn[p] * features[b,p,:] (f32, coalesced) ----
  float4 cv = make_float4(0.f, 0.f, 0.f, 0.f);
#pragma unroll 8
  for (int p = 0; p < 64; ++p) {
    const float a = sAttn[p];
    const float4 v = f4[p * 512 + tid];
    cv.x += a * v.x;  cv.y += a * v.y;  cv.z += a * v.z;  cv.w += a * v.w;
  }
  ((float4*)(out_ctx + (size_t)b * 2048))[tid] = cv;
}

extern "C" void kernel_launch(void* const* d_in, const int* in_sizes, int n_in,
                              void* d_out, int out_size, void* d_ws, size_t ws_size,
                              hipStream_t stream) {
  const float* features = (const float*)d_in[0];
  const float* hidden   = (const float*)d_in[1];
  const float* W1w      = (const float*)d_in[2];
  const float* W1b      = (const float*)d_in[3];
  const float* W2w      = (const float*)d_in[4];
  const float* W2b      = (const float*)d_in[5];
  const float* Vw       = (const float*)d_in[6];
  // d_in[7] = V_b : uniform logit shift, softmax-invariant -> unused.

  char* ws = (char*)d_ws;
  unsigned short* W1T = (unsigned short*)(ws);                             // 4 MB
  unsigned short* W2T = (unsigned short*)(ws + 4u * 1024 * 1024);          // 2 MB
  unsigned short* HB  = (unsigned short*)(ws + 6u * 1024 * 1024);          // 0.5 MB
  float*          PH  = (float*)(ws + 6u * 1024 * 1024 + 512u * 1024);     // 1 MB

  float* out_ctx  = (float*)d_out;           // 256*2048
  float* out_attn = out_ctx + 256 * 2048;    // 256*64

  const dim3 tb(32, 8);
  k_transpose_bf16<<<dim3(2048 / 32, 1024 / 32), tb, 0, stream>>>(W1w, W1T, 2048, 1024);
  k_transpose_bf16<<<dim3(1024 / 32, 1024 / 32), tb, 0, stream>>>(W2w, W2T, 1024, 1024);
  k_f32_to_bf16<<<(256 * 1024) / 256, 256, 0, stream>>>(hidden, HB, 256 * 1024);
  k_proj_h<<<dim3(16, 16), 128, 0, stream>>>(HB, W2T, W2b, PH);

  const size_t smem = 64 * 2048 * 2 + 128 * sizeof(float);  // 262656 B (< 320 KB/WGP)
  k_attention<<<256, 512, smem, stream>>>(features, W1T, W1b, Vw, PH, out_ctx, out_attn);
}
